// HyperGraphAttentionLayerSparse_57329223467439
// MI455X (gfx1250) — matrix operand order repaired and checked by reading, not compiled, not run on hardware
//
#include <hip/hip_runtime.h>
#include <stdint.h>

#define N2 10000
#define N1 10000
#define FD 256
#define ALPHA 0.2f

#define BM 128
#define BN 128
#define BK 32
#define LSTR 40            // LDS row stride (bf16 elems) for A tile: 80B rows, 16B aligned, conflict-free
#define KP 10048           // padded K stride for transposed B operands (157*64), zero-filled tail
#define TP 72              // transpose-kernel LDS stride (144B rows, 16B aligned)

typedef __bf16 bf16;
typedef __attribute__((ext_vector_type(16))) __bf16 v16bf;
typedef __attribute__((ext_vector_type(8)))  float  v8f;

__device__ __forceinline__ bf16 to_bf(float f) {
  unsigned u = __builtin_bit_cast(unsigned, f);
  unsigned r = u + 0x7FFFu + ((u >> 16) & 1u);
  unsigned short h = (unsigned short)(r >> 16);
  return __builtin_bit_cast(bf16, h);
}

__device__ __forceinline__ float lrelu(float x) { return x > 0.f ? x : ALPHA * x; }

__device__ __forceinline__ float wave_sum(float v) {
#pragma unroll
  for (int m = 16; m >= 1; m >>= 1) v += __shfl_xor(v, m, 32);
  return v;
}

// A fragment (16 rows x 32 K): lanes 0-15 hold K {0-7,16-23}; lanes 16-31 hold K {8-15,24-31}
__device__ __forceinline__ v16bf fragA(const bf16* base, int row0, int stride, int kofs, int lane) {
  const bf16* p = base + (size_t)(row0 + (lane & 15)) * stride + kofs + ((lane >> 4) << 3);
  union { v16bf v; uint4 q[2]; } u;
  u.q[0] = *(const uint4*)(p);
  u.q[1] = *(const uint4*)(p + 16);
  return u.v;
}

// B fragment (32 K x 16 cols), source stored column-major [n][k]: lanes 0-15 hold K {0-7,16-23}; lanes 16-31 hold K {8-15,24-31}
__device__ __forceinline__ v16bf fragB(const bf16* base, int col0, int stride, int kofs, int lane) {
  const bf16* p = base + (size_t)(col0 + (lane & 15)) * stride + kofs + ((lane >> 4) << 3);
  union { v16bf v; uint4 q[2]; } u;
  u.q[0] = *(const uint4*)(p);
  u.q[1] = *(const uint4*)(p + 16);
  return u.v;
}

// W-value pack: adj gate * exp(lrelu(sn + se)) -> bf16 (select keeps zeros exact, discards exp of garbage)
__device__ __forceinline__ void pack4(bf16* h, float4 a, float4 se, float sni) {
  h[0] = to_bf(a.x != 0.f ? a.x * __expf(lrelu(sni + se.x)) : 0.f);
  h[1] = to_bf(a.y != 0.f ? a.y * __expf(lrelu(sni + se.y)) : 0.f);
  h[2] = to_bf(a.z != 0.f ? a.z * __expf(lrelu(sni + se.z)) : 0.f);
  h[3] = to_bf(a.w != 0.f ? a.w * __expf(lrelu(sni + se.w)) : 0.f);
}

// ---------------- prep kernels ----------------

__global__ void k_convert_x(const float* __restrict__ x, bf16* __restrict__ xb, int n) {
  int i = blockIdx.x * blockDim.x + threadIdx.x;
  if (i < n) xb[i] = to_bf(x[i]);
}

// wt[n][k] = w[k][n]
__global__ void k_prep_w(const float* __restrict__ w, bf16* __restrict__ wt) {
  int n = blockIdx.x;
  int k = threadIdx.x;
  wt[n * FD + k] = to_bf(w[k * FD + n]);
}

// w3a[k] = sum_n weight3[k][n]*a2[FD+n];  c0 = wc . a[:FD]
__global__ void k_w3a(const float* __restrict__ w3, const float* __restrict__ a2,
                      const float* __restrict__ a, const float* __restrict__ wc,
                      float* __restrict__ w3a, float* __restrict__ c0) {
  int k = threadIdx.x;
  float s = 0.f;
  for (int n = 0; n < FD; ++n) s += w3[k * FD + n] * a2[FD + n];
  w3a[k] = s;
  if (k == 0) {
    float c = 0.f;
    for (int t = 0; t < FD; ++t) c += wc[t] * a[t];
    *c0 = c;
  }
}

// tiled bf16 transpose: src[M][256] -> dst[256][KP], rows M..KP-1 zero-filled
__global__ void __launch_bounds__(256) k_transpose_bf(const bf16* __restrict__ src,
                                                      bf16* __restrict__ dst, int M) {
  __shared__ __align__(16) bf16 t[64 * TP];
  int bi = blockIdx.x;   // 157 source-row tiles (covers KP)
  int bj = blockIdx.y;   // 4 source-col tiles
  int tid = threadIdx.x;
#pragma unroll
  for (int p = 0; p < 2; ++p) {
    int lin = p * 256 + tid;
    int row = lin >> 3;
    int g = lin & 7;
    union { uint4 q; bf16 h[8]; } u;
    u.q = make_uint4(0u, 0u, 0u, 0u);
    int gr = bi * 64 + row;
    if (gr < M) u.q = *(const uint4*)(src + (size_t)gr * FD + bj * 64 + g * 8);
    *(uint4*)(&t[row * TP + g * 8]) = u.q;
  }
  __syncthreads();
#pragma unroll
  for (int p = 0; p < 2; ++p) {
    int lin = p * 256 + tid;
    int c = lin >> 3;       // local col -> dst row
    int g = lin & 7;        // group of 8 along dst K
    union { uint4 q; bf16 h[8]; } u;
#pragma unroll
    for (int q = 0; q < 8; ++q) u.h[q] = t[(g * 8 + q) * TP + c];
    *(uint4*)(dst + (size_t)(bj * 64 + c) * KP + bi * 64 + g * 8) = u.q;
  }
}

// ---------------- small GEMM: C[10000][256] = A_bf[10000][256] @ Bt_bf[256n][256k]^T ----------------
__global__ void __launch_bounds__(256) k_gemm_small(const bf16* __restrict__ A,
                                                    const bf16* __restrict__ Bt,
                                                    float* __restrict__ C) {
  int lane = threadIdx.x & 31;
  int wid = (blockIdx.x * blockDim.x + threadIdx.x) >> 5;
  int mt = wid >> 4;
  int nt = wid & 15;
  if (mt >= N2 / 16) return;
  int m0 = mt * 16, n0 = nt * 16;
  v8f acc = {};
#pragma unroll
  for (int k = 0; k < FD; k += BK) {
    v16bf af = fragA(A,  m0, FD, k, lane);
    v16bf bq = fragB(Bt, n0, FD, k, lane);
    acc = __builtin_amdgcn_wmma_f32_16x16x32_bf16(false, af, false, bq, (short)0, acc, false, false);
  }
  int rbase = m0 + ((lane >> 4) << 3);
  int col = n0 + (lane & 15);
#pragma unroll
  for (int r = 0; r < 8; ++r) C[(size_t)(rbase + r) * FD + col] = acc[r];
}

// ---------------- per-row stats ----------------
__global__ void k_rowstats(const float* __restrict__ x4att, const float* __restrict__ xw,
                           const float* __restrict__ aTail, const float* __restrict__ a2Head,
                           const float* __restrict__ c0p, float* __restrict__ g,
                           float* __restrict__ s_node, bf16* __restrict__ xtg) {
  int lane = threadIdx.x & 31;
  int wid = (blockIdx.x * blockDim.x + threadIdx.x) >> 5;
  int nw = (gridDim.x * blockDim.x) >> 5;
  float c0 = *c0p;
  for (int i = wid; i < N2; i += nw) {
    const float* row = x4att + (size_t)i * FD;
    float s1 = 0.f, sn = 0.f;
#pragma unroll
    for (int k = lane; k < FD; k += 32) {
      float v = row[k];
      s1 += v * aTail[k];
      sn += v * a2Head[k];
    }
    s1 = wave_sum(s1);
    sn = wave_sum(sn);
    float gi = __expf(lrelu(s1 + c0));
    if (lane == 0) { g[i] = gi; s_node[i] = sn; }
    const float* xr = xw + (size_t)i * FD;
    bf16* o = xtg + (size_t)i * FD;
#pragma unroll
    for (int k = lane; k < FD; k += 32) o[k] = to_bf(gi * xr[k]);
  }
}

// ---------------- Z1[j] = sum_i adj[i][j]*g[i] ----------------
__global__ void k_z1(const float* __restrict__ adj, const float* __restrict__ g,
                     float* __restrict__ Z1) {
  int j = blockIdx.x * blockDim.x + threadIdx.x;
  if (j >= N1) return;
  float s = 0.f;
  int i = 0;
  for (; i + 4 <= N2; i += 4) {
    if (i + 32 < N2) __builtin_prefetch(adj + (size_t)(i + 32) * N1 + j, 0, 1);
    s += adj[(size_t)(i + 0) * N1 + j] * g[i + 0];
    s += adj[(size_t)(i + 1) * N1 + j] * g[i + 1];
    s += adj[(size_t)(i + 2) * N1 + j] * g[i + 2];
    s += adj[(size_t)(i + 3) * N1 + j] * g[i + 3];
  }
  for (; i < N2; ++i) s += adj[(size_t)i * N1 + j] * g[i];
  Z1[j] = s;
}

// ---------------- stage-1 big GEMM: edge[j][f] = (sum_i adj[i][j]*xtg[i][f]) / Z1[j] ----------------
// A (adj^T) via LDS transpose+cvt; B read directly from xtgT[256][KP] (global, L2-resident).
__global__ void __launch_bounds__(256) k_edge_gemm(const float* __restrict__ adj,
                                                   const bf16* __restrict__ xtgT,
                                                   const float* __restrict__ Z1,
                                                   float* __restrict__ edge,
                                                   bf16* __restrict__ edge_bf) {
  __shared__ __align__(16) bf16 Asm[BM * LSTR];
  int tid = threadIdx.x;
  int lane = tid & 31;
  int wv = tid >> 5;
  int wm = wv & 3, wn = wv >> 2;
  int j0 = blockIdx.y * BM;
  int n0 = blockIdx.x * BN;
  v8f acc[2][4] = {};
  int ksteps = (N2 + BK - 1) / BK;
  for (int ks = 0; ks < ksteps; ++ks) {
    int k0 = ks * BK;
    __syncthreads();
#pragma unroll
    for (int p = 0; p < 4; ++p) {
      int lin = p * 256 + tid;
      int r = lin >> 5;        // i offset 0..31
      int c = lin & 31;        // float4 group along j
      int gi = k0 + r;
      int gj = j0 + c * 4;
      float4 v = make_float4(0.f, 0.f, 0.f, 0.f);
      if (gi < N2 && gj + 3 < N1) v = *(const float4*)(adj + (size_t)gi * N1 + gj);
      Asm[(c * 4 + 0) * LSTR + r] = to_bf(v.x);
      Asm[(c * 4 + 1) * LSTR + r] = to_bf(v.y);
      Asm[(c * 4 + 2) * LSTR + r] = to_bf(v.z);
      Asm[(c * 4 + 3) * LSTR + r] = to_bf(v.w);
    }
    __syncthreads();
    v16bf af[2], bq[4];
#pragma unroll
    for (int i = 0; i < 2; ++i) af[i] = fragA(Asm, wm * 32 + i * 16, LSTR, 0, lane);
#pragma unroll
    for (int i = 0; i < 4; ++i) bq[i] = fragB(xtgT, n0 + wn * 64 + i * 16, KP, k0, lane);
#pragma unroll
    for (int i = 0; i < 2; ++i)
#pragma unroll
      for (int jx = 0; jx < 4; ++jx)
        acc[i][jx] = __builtin_amdgcn_wmma_f32_16x16x32_bf16(false, af[i], false, bq[jx],
                                                             (short)0, acc[i][jx], false, false);
  }
  int rbase = wm * 32 + ((lane >> 4) << 3);
  int cbase = wn * 64 + (lane & 15);
#pragma unroll
  for (int i = 0; i < 2; ++i) {
#pragma unroll
    for (int jx = 0; jx < 4; ++jx) {
      int col = n0 + cbase + jx * 16;
#pragma unroll
      for (int r = 0; r < 8; ++r) {
        int row = j0 + rbase + i * 16 + r;
        if (row < N1) {
          float val = acc[i][jx][r] / Z1[row];
          edge[(size_t)row * FD + col] = val;
          edge_bf[(size_t)row * FD + col] = to_bf(val);
        }
      }
    }
  }
}

// ---------------- s_edge[j] = edge[j][:] . w3a ----------------
__global__ void k_sedge(const float* __restrict__ edge, const float* __restrict__ w3a,
                        float* __restrict__ s_edge) {
  int lane = threadIdx.x & 31;
  int wid = (blockIdx.x * blockDim.x + threadIdx.x) >> 5;
  int nw = (gridDim.x * blockDim.x) >> 5;
  for (int j = wid; j < N1; j += nw) {
    const float* row = edge + (size_t)j * FD;
    float s = 0.f;
#pragma unroll
    for (int k = lane; k < FD; k += 32) s += row[k] * w3a[k];
    s = wave_sum(s);
    if (lane == 0) s_edge[j] = s;
  }
}

// ---------------- Z2[i] = sum_j adj[i][j]*exp(lrelu(s_node[i]+s_edge[j])) ----------------
__global__ void k_z2(const float* __restrict__ adj, const float* __restrict__ s_node,
                     const float* __restrict__ s_edge, float* __restrict__ Z2) {
  int lane = threadIdx.x & 31;
  int wid = (blockIdx.x * blockDim.x + threadIdx.x) >> 5;
  if (wid >= N2) return;
  float sn = s_node[wid];
  const float* row = adj + (size_t)wid * N1;
  float s = 0.f;
  for (int j = lane; j < N1; j += 32) {
    float aij = row[j];
    if (aij != 0.f) s += aij * __expf(lrelu(sn + s_edge[j]));
  }
  s = wave_sum(s);
  if (lane == 0) Z2[wid] = s;
}

// ---------------- stage-2 big GEMM, zero-LDS zero-barrier ----------------
// A built per-lane in registers: W[i][j] = adj[i][j]*exp(lrelu(sn_i+se_j)); B direct from edgeT[256][KP].
__global__ void __launch_bounds__(256) k_node_gemm(const float* __restrict__ adj,
                                                   const bf16* __restrict__ edgeT,
                                                   const float* __restrict__ s_node,
                                                   const float* __restrict__ s_edge,
                                                   const float* __restrict__ Z2,
                                                   float* __restrict__ out) {
  int tid = threadIdx.x;
  int lane = tid & 31;
  int wv = tid >> 5;
  int wm = wv & 3, wn = wv >> 2;
  int i0 = blockIdx.y * BM;
  int n0 = blockIdx.x * BN;
  int half = lane >> 4;
  int l15 = lane & 15;
  int mrow[2]; float sn[2];
#pragma unroll
  for (int i = 0; i < 2; ++i) {
    mrow[i] = i0 + wm * 32 + i * 16 + l15;
    sn[i] = (mrow[i] < N2) ? s_node[mrow[i]] : 0.f;
  }
  v8f acc[2][4] = {};
  int ksteps = (N1 + BK - 1) / BK;
  for (int ks = 0; ks < ksteps; ++ks) {
    int k0 = ks * BK;
    int kb = k0 + half * 8;
    // s_edge reads stay inside padded allocation; tail garbage is discarded by the adj gate
    float4 se0 = *(const float4*)(s_edge + kb);
    float4 se1 = *(const float4*)(s_edge + kb + 4);
    float4 se2 = *(const float4*)(s_edge + kb + 16);
    float4 se3 = *(const float4*)(s_edge + kb + 20);
    v16bf af[2];
#pragma unroll
    for (int i = 0; i < 2; ++i) {
      const float* ap = adj + (size_t)mrow[i] * N1 + kb;
      float4 z = make_float4(0.f, 0.f, 0.f, 0.f);
      float4 a0 = z, a1 = z, a2v = z, a3 = z;
      bool rok = (mrow[i] < N2);
      if (rok && kb + 3  < N1) a0  = *(const float4*)(ap);
      if (rok && kb + 7  < N1) a1  = *(const float4*)(ap + 4);
      if (rok && kb + 19 < N1) a2v = *(const float4*)(ap + 16);
      if (rok && kb + 23 < N1) a3  = *(const float4*)(ap + 20);
      union { v16bf v; bf16 h[16]; } uf;
      pack4(uf.h + 0,  a0,  se0, sn[i]);
      pack4(uf.h + 4,  a1,  se1, sn[i]);
      pack4(uf.h + 8,  a2v, se2, sn[i]);
      pack4(uf.h + 12, a3,  se3, sn[i]);
      af[i] = uf.v;
    }
    v16bf bq[4];
#pragma unroll
    for (int jx = 0; jx < 4; ++jx) bq[jx] = fragB(edgeT, n0 + wn * 64 + jx * 16, KP, k0, lane);
#pragma unroll
    for (int i = 0; i < 2; ++i)
#pragma unroll
      for (int jx = 0; jx < 4; ++jx)
        acc[i][jx] = __builtin_amdgcn_wmma_f32_16x16x32_bf16(false, af[i], false, bq[jx],
                                                             (short)0, acc[i][jx], false, false);
  }
  int rbase = wm * 32 + ((lane >> 4) << 3);
  int cbase = wn * 64 + (lane & 15);
#pragma unroll
  for (int i = 0; i < 2; ++i) {
#pragma unroll
    for (int jx = 0; jx < 4; ++jx) {
      int col = n0 + cbase + jx * 16;
#pragma unroll
      for (int r = 0; r < 8; ++r) {
        int row = i0 + rbase + i * 16 + r;
        if (row < N2) {
          float val = acc[i][jx][r] / Z2[row];
          out[(size_t)row * FD + col] = val > 0.f ? val : 0.f;
        }
      }
    }
  }
}

// ---------------- launch ----------------
extern "C" void kernel_launch(void* const* d_in, const int* in_sizes, int n_in,
                              void* d_out, int out_size, void* d_ws, size_t ws_size,
                              hipStream_t stream) {
  (void)in_sizes; (void)n_in; (void)out_size; (void)ws_size;
  const float* x       = (const float*)d_in[0];
  const float* adj     = (const float*)d_in[1];
  const float* weight  = (const float*)d_in[2];
  const float* weight2 = (const float*)d_in[3];
  const float* weight3 = (const float*)d_in[4];
  const float* a       = (const float*)d_in[5];
  const float* a2      = (const float*)d_in[6];
  const float* wc      = (const float*)d_in[7];
  float* out = (float*)d_out;

  char* ws = (char*)d_ws;
  size_t off = 0;
  auto alloc = [&](size_t bytes) -> void* {
    off = (off + 255) & ~(size_t)255;
    void* p = ws + off;
    off += bytes;
    return p;
  };
  bf16*  x_bf   = (bf16*)alloc((size_t)N2 * FD * 2);
  bf16*  w2t    = (bf16*)alloc((size_t)FD * FD * 2);
  bf16*  wt     = (bf16*)alloc((size_t)FD * FD * 2);
  float* w3a    = (float*)alloc(FD * 4);
  float* c0     = (float*)alloc(256);
  float* x4att  = (float*)alloc((size_t)N2 * FD * 4);
  float* xw     = (float*)alloc((size_t)N2 * FD * 4);
  bf16*  xtg    = (bf16*)alloc((size_t)N2 * FD * 2);
  bf16*  xtgT   = (bf16*)alloc((size_t)FD * KP * 2);
  float* g      = (float*)alloc((size_t)N2 * 4);
  float* s_node = (float*)alloc((size_t)N2 * 4);
  float* Z1     = (float*)alloc((size_t)N1 * 4);
  float* s_edge = (float*)alloc((size_t)(N1 + 64) * 4);   // padded for lane-indexed tail reads
  float* Z2     = (float*)alloc((size_t)N2 * 4);
  float* edge   = (float*)alloc((size_t)N1 * FD * 4);
  bf16*  edge_bf= (bf16*)alloc((size_t)N1 * FD * 2);
  bf16*  edgeT  = (bf16*)alloc((size_t)FD * KP * 2);

  k_convert_x<<<(N2 * FD + 255) / 256, 256, 0, stream>>>(x, x_bf, N2 * FD);
  k_prep_w<<<FD, FD, 0, stream>>>(weight2, w2t);
  k_prep_w<<<FD, FD, 0, stream>>>(weight, wt);
  k_w3a<<<1, FD, 0, stream>>>(weight3, a2, a, wc, w3a, c0);

  k_gemm_small<<<1250, 256, 0, stream>>>(x_bf, w2t, x4att);   // x @ weight2
  k_gemm_small<<<1250, 256, 0, stream>>>(x_bf, wt, xw);       // x @ weight

  k_rowstats<<<512, 256, 0, stream>>>(x4att, xw, a + FD, a2, c0, g, s_node, xtg);

  dim3 gtr(KP / 64, FD / 64);   // (157, 4)
  k_transpose_bf<<<gtr, 256, 0, stream>>>(xtg, xtgT, N2);

  k_z1<<<(N1 + 255) / 256, 256, 0, stream>>>(adj, g, Z1);

  dim3 gbig(FD / BN, (N1 + BM - 1) / BM);   // (2, 79)
  k_edge_gemm<<<gbig, 256, 0, stream>>>(adj, xtgT, Z1, edge, edge_bf);

  k_transpose_bf<<<gtr, 256, 0, stream>>>(edge_bf, edgeT, N1);

  k_sedge<<<512, 256, 0, stream>>>(edge, w3a, s_edge);
  k_z2<<<(N2 * 32 + 255) / 256, 256, 0, stream>>>(adj, s_node, s_edge, Z2);

  dim3 gbig2(FD / BN, (N2 + BM - 1) / BM);
  k_node_gemm<<<gbig2, 256, 0, stream>>>(adj, edgeT, s_node, s_edge, Z2, out);
}
